// ClipMatcher_13108240187771
// MI455X (gfx1250) — matrix operand order repaired and checked by reading, not compiled, not run on hardware
//
#include <hip/hip_runtime.h>
#include <cstdint>

// ---------------------------------------------------------------------------
// MI455X (gfx1250, wave32) anchor-matching loss.
// Nine 1x1-conv layers as bf16 WMMA GEMMs (v_wmma_f32_16x16x32_bf16), BN
// folded into weights, bias+LeakyReLU fused. Activations are stored
// position-major as packed bf16 channel-pairs [b][l][c/2] so that
//  - LDS tiles are filled with CDNA5 global_load_async_to_lds_b128
//  - A- and B-fragments are contiguous in LDS -> ds_load_b128 only
//  - epilogue writes one b128 per 16x16 subtile per lane
// feat (f32 planar) is transpose-converted once by a small tile kernel.
// ---------------------------------------------------------------------------

typedef __attribute__((ext_vector_type(16))) __bf16 v16bf;
typedef __attribute__((ext_vector_type(8)))  float  v8f;
typedef int v4i_ __attribute__((vector_size(16)));   // matches builtin param type

#define LDIM 256          // L (sequence length) == K (channels) == 256
#define KDW  128          // K/2 packed-bf16 dwords per weight row
#define XSTR 20           // X LDS row stride ([l][k2] layout, 16B-aligned)
#define WSTR 20           // W LDS row stride ([o][k2] layout, 16B-aligned)

#if defined(__HIP_DEVICE_COMPILE__) && \
    __has_builtin(__builtin_amdgcn_global_load_async_to_lds_b128) && \
    __has_builtin(__builtin_amdgcn_s_wait_asynccnt)
#define HAS_ASYNC 1
#define GA1(p) ((__attribute__((address_space(1))) v4i_*)(p))
#define LA3(p) ((__attribute__((address_space(3))) v4i_*)(p))
#else
#define HAS_ASYNC 0
#endif

__device__ __forceinline__ unsigned bf16_rne(float x) {
  unsigned u = __float_as_uint(x);
  return (u + 0x7fffu + ((u >> 16) & 1u)) >> 16;
}
__device__ __forceinline__ unsigned f2bf_pack(float a, float b) {
  return (bf16_rne(a) & 0xffffu) | (bf16_rne(b) << 16);
}

// ---------------------------------------------------------------------------
// Fold BN into conv weights, pack bf16 pairs [o][k2] (low16 = even k).
// ---------------------------------------------------------------------------
__global__ __launch_bounds__(256)
void prep_weights(const float* __restrict__ w, const float* __restrict__ b,
                  const float* __restrict__ g, const float* __restrict__ bt,
                  const float* __restrict__ rm, const float* __restrict__ rv,
                  int Nact, int Npad,
                  unsigned* __restrict__ W2, float* __restrict__ bout)
{
  int i = blockIdx.x * 256 + threadIdx.x;
  if (i >= Npad * KDW) return;
  int o = i >> 7, k2 = i & 127;
  float s = 1.0f, bb = 0.0f, w0 = 0.0f, w1 = 0.0f;
  if (o < Nact) {
    if (g != nullptr) {
      s  = g[o] * rsqrtf(rv[o] + 1e-5f);
      bb = s * (b[o] - rm[o]) + bt[o];
    } else {
      bb = b[o];
    }
    w0 = s * w[o * 256 + 2 * k2];
    w1 = s * w[o * 256 + 2 * k2 + 1];
  }
  W2[i] = f2bf_pack(w0, w1);
  if (k2 == 0) bout[o] = bb;
}

// ---------------------------------------------------------------------------
// feat [b][c][l] f32  ->  packed bf16 pairs, position-major [b][l][c/2]
// 64x64 tile through LDS: coalesced f32 reads, b128 packed writes.
// ---------------------------------------------------------------------------
__global__ __launch_bounds__(256)
void convert_feat(const float* __restrict__ feat, unsigned* __restrict__ outp)
{
  __shared__ float T[64][65];
  const int b   = blockIdx.z;
  const int l0  = blockIdx.x << 6;   // 0..192
  const int c0  = blockIdx.y << 6;   // 0..192
  const int tid = threadIdx.x;

  { // load 64(c) x 64(l), coalesced along l
    const int c  = tid >> 2;         // 0..63
    const int lq = (tid & 3) << 4;   // 0,16,32,48
    const float* src = feat + ((long)b * 256 + (c0 + c)) * 256 + l0 + lq;
#pragma unroll
    for (int q = 0; q < 4; ++q) {
      float4 u = *(const float4*)(src + 4 * q);
      T[c][lq + 4 * q + 0] = u.x;
      T[c][lq + 4 * q + 1] = u.y;
      T[c][lq + 4 * q + 2] = u.z;
      T[c][lq + 4 * q + 3] = u.w;
    }
  }
  __syncthreads();
  { // pack (c even, c odd) pairs along c for one l position
    const int l  = tid >> 2;         // 0..63
    const int cg = (tid & 3) << 3;   // c2 offset 0,8,16,24
    uint4 p0, p1;
    unsigned* p = (unsigned*)&p0;
#pragma unroll
    for (int q = 0; q < 4; ++q)
      p[q] = f2bf_pack(T[2 * (cg + q)][l], T[2 * (cg + q) + 1][l]);
    p = (unsigned*)&p1;
#pragma unroll
    for (int q = 0; q < 4; ++q)
      p[q] = f2bf_pack(T[2 * (cg + 4 + q)][l], T[2 * (cg + 4 + q) + 1][l]);
    unsigned* d = outp + ((long)b * 256 + l0 + l) * 128 + (c0 >> 1) + cg;
    *(uint4*)(d)     = p0;
    *(uint4*)(d + 4) = p1;
  }
}

// ---------------------------------------------------------------------------
// WMMA GEMM: out[b][l][o] = act(sum_c W[o][c] * in[b][l][c] + bias[o])
//   block 128 thr (4 waves 2x2) -> tile 64(o) x 128(l); wave 32x64 = 8 WMMA
//   per 32-wide K chunk; K = 256 = 8 chunks; LDS double-buffered via async
//   global->LDS b128 copies, s_wait_asynccnt pipelining.
//   Packed-in : dwords [b][l][c/2]  (pair = bf16(c even)|bf16(c odd)<<16)
//   Packed-out: dwords [b][l][o/2] ; f32-out: floats [b][l][o]
// ---------------------------------------------------------------------------
template <bool OUT_F32>
__global__ __launch_bounds__(128)
void gemm_bn_lrelu(const unsigned* __restrict__ inp, long inBS, int inRowDW,
                   const unsigned* __restrict__ W2, const float* __restrict__ bias,
                   void* __restrict__ outp, long outBS, int outRowDW)
{
  __shared__ unsigned Xl[2][128 * XSTR];
  __shared__ unsigned Wl[2][64 * WSTR];

  const int tid  = threadIdx.x;
  const int mt   = blockIdx.x;          // 0..479 : (b, l/128)
  const int b    = mt >> 1;
  const int l0   = (mt & 1) << 7;
  const int o0   = blockIdx.y << 6;

  const int wave = tid >> 5, lane = tid & 31;
  const int half = lane >> 4, l16 = lane & 15;
  const int wo   = wave >> 1;           // wave o-half  (32 rows)
  const int wm   = wave & 1;            // wave l-half  (64 cols)

  const int wrow = tid >> 1, wch = (tid & 1) << 3;   // W fill: row 0..63, col 0/8

  v8f acc[2][4];
#pragma unroll
  for (int i = 0; i < 2; ++i)
#pragma unroll
    for (int s = 0; s < 4; ++s) {
      v8f z = {0.f, 0.f, 0.f, 0.f, 0.f, 0.f, 0.f, 0.f};
      acc[i][s] = z;
    }

  auto issue = [&](int bi, int kc) {
    { // weight tile -> Wl[bi][o][k2]
      const unsigned* g = W2 + (long)(o0 + wrow) * KDW + kc * 16 + wch;
      unsigned* d = &Wl[bi][wrow * WSTR + wch];
#if HAS_ASYNC
      __builtin_amdgcn_global_load_async_to_lds_b128(GA1(g),     LA3(d),     0, 0);
      __builtin_amdgcn_global_load_async_to_lds_b128(GA1(g + 4), LA3(d + 4), 0, 0);
#else
#pragma unroll
      for (int q = 0; q < 8; ++q) d[q] = g[q];
#endif
    }
    { // activation tile: one l-row (16 dwords) per thread -> Xl[bi][l][k2]
      const unsigned* g = inp + (long)b * inBS + (long)(l0 + tid) * inRowDW + kc * 16;
      unsigned* d = &Xl[bi][tid * XSTR];
#if HAS_ASYNC
#pragma unroll
      for (int q = 0; q < 4; ++q)
        __builtin_amdgcn_global_load_async_to_lds_b128(GA1(g + 4 * q), LA3(d + 4 * q), 0, 0);
#else
#pragma unroll
      for (int q = 0; q < 16; ++q) d[q] = g[q];
#endif
    }
  };

  issue(0, 0);
  for (int kc = 0; kc < 8; ++kc) {
    const int bi = kc & 1;
    if (kc < 7) issue(bi ^ 1, kc + 1);
#if HAS_ASYNC
    // in-order async completion: <=6 outstanding == current chunk resident
    if (kc < 7) __builtin_amdgcn_s_wait_asynccnt(6);
    else        __builtin_amdgcn_s_wait_asynccnt(0);
#endif
    __syncthreads();

    union Frag { unsigned u[8]; v16bf v; };
    Frag a0, a1;
#pragma unroll
    for (int j = 0; j < 8; ++j) {
      const int k2 = (j + (j & 4)) + (half << 2);  // A: {0..3,8..11} + 4*half
      a0.u[j] = Wl[bi][((wo << 5) + l16) * WSTR + k2];
      a1.u[j] = Wl[bi][((wo << 5) + 16 + l16) * WSTR + k2];
    }
#pragma unroll
    for (int s = 0; s < 4; ++s) {
      Frag bx;
      const int row = (wm << 6) + (s << 4) + l16;
#pragma unroll
      for (int j = 0; j < 8; ++j)                   // B: {0..3,8..11} + 4*half
        bx.u[j] = Xl[bi][row * XSTR + (j + (j & 4)) + (half << 2)];
      acc[0][s] = __builtin_amdgcn_wmma_f32_16x16x32_bf16(
          false, a0.v, false, bx.v, (short)0, acc[0][s], false, false);
      acc[1][s] = __builtin_amdgcn_wmma_f32_16x16x32_bf16(
          false, a1.v, false, bx.v, (short)0, acc[1][s], false, false);
    }
    __syncthreads();
  }

  // epilogue: D element (VGPR v, lane) = (M = o = v + 8*half, N = l = l16)
#pragma unroll
  for (int i = 0; i < 2; ++i) {
    const int ob = o0 + (wo << 5) + (i << 4);
#pragma unroll
    for (int s = 0; s < 4; ++s) {
      const int lb = l0 + (wm << 6) + (s << 4) + l16;
      if constexpr (OUT_F32) {
        float4 f0, f1;
        f0.x = acc[i][s][0] + bias[ob + (half << 3) + 0];
        f0.y = acc[i][s][1] + bias[ob + (half << 3) + 1];
        f0.z = acc[i][s][2] + bias[ob + (half << 3) + 2];
        f0.w = acc[i][s][3] + bias[ob + (half << 3) + 3];
        f1.x = acc[i][s][4] + bias[ob + (half << 3) + 4];
        f1.y = acc[i][s][5] + bias[ob + (half << 3) + 5];
        f1.z = acc[i][s][6] + bias[ob + (half << 3) + 6];
        f1.w = acc[i][s][7] + bias[ob + (half << 3) + 7];
        float* d = (float*)outp + (long)b * outBS + (long)lb * outRowDW + ob + (half << 3);
        *(float4*)(d)     = f0;
        *(float4*)(d + 4) = f1;
      } else {
        uint4 pk;
        unsigned* p = (unsigned*)&pk;
#pragma unroll
        for (int k = 0; k < 4; ++k) {
          const int o = ob + 2 * k + (half << 3);
          float v0 = acc[i][s][2 * k]     + bias[o];
          float v1 = acc[i][s][2 * k + 1] + bias[o + 1];
          v0 = v0 > 0.f ? v0 : 0.01f * v0;
          v1 = v1 > 0.f ? v1 : 0.01f * v1;
          p[k] = f2bf_pack(v0, v1);
        }
        unsigned* d = (unsigned*)outp + (long)b * outBS + (long)lb * outRowDW +
                      (ob >> 1) + (half << 2);
        *(uint4*)(d) = pk;
      }
    }
  }
}

// ---------------------------------------------------------------------------
// Loss: anchor gen + IoU match + L1/GIoU/BCE, wave-reduce + global atomics.
// ---------------------------------------------------------------------------
__device__ __forceinline__ void iou_giou_dev(float ax1, float ay1, float ax2, float ay2,
                                             float bx1, float by1, float bx2, float by2,
                                             float& iou, float& giou)
{
  float x1 = fmaxf(ax1, bx1), y1 = fmaxf(ay1, by1);
  float x2 = fminf(ax2, bx2), y2 = fminf(ay2, by2);
  float inter = fmaxf(x2 - x1, 0.f) * fmaxf(y2 - y1, 0.f);
  float aa = (ax2 - ax1) * (ay2 - ay1);
  float ab = (bx2 - bx1) * (by2 - by1);
  float un = aa + ab - inter;
  iou = inter / (un + 1e-7f);
  float ex1 = fminf(ax1, bx1), ey1 = fminf(ay1, by1);
  float ex2 = fmaxf(ax2, bx2), ey2 = fmaxf(ay2, by2);
  float enc = (ex2 - ex1) * (ey2 - ey1);
  giou = iou - (enc - un) / (enc + 1e-7f);
}

__global__ __launch_bounds__(256)
void loss_kernel(const float* __restrict__ oreg, const float* __restrict__ ocls,
                 const float* __restrict__ gt, float* __restrict__ sums)
{
  const long gid = (long)blockIdx.x * 256 + threadIdx.x;   // 0 .. 240*3072-1
  const int b = (int)(gid / 3072);
  const int a = (int)(gid % 3072);
  const int l = a / 12, j = a % 12, n = j / 3, m = j % 3;

  const float sq[3] = {0.70710678118654752f, 1.0f, 1.41421356237309505f};
  const float base = (float)(16 << n);
  const float ax = ((l & 15) + 0.5f) * (1.0f / 16.0f);
  const float ay = ((l >> 4) + 0.5f) * (1.0f / 16.0f);
  const float aw = base / (sq[m] * 448.0f);
  const float ah = base * sq[m] * (1.0f / 448.0f);

  const float gx = gt[b * 4 + 0], gy = gt[b * 4 + 1];
  const float gw = gt[b * 4 + 2], gh = gt[b * 4 + 3];
  const float gx1 = gx - 0.5f * gw, gy1 = gy - 0.5f * gh;
  const float gx2 = gx + 0.5f * gw, gy2 = gy + 0.5f * gh;

  float iou, gdum;
  iou_giou_dev(ax - 0.5f * aw, ay - 0.5f * ah, ax + 0.5f * aw, ay + 0.5f * ah,
               gx1, gy1, gx2, gy2, iou, gdum);
  const float pos = iou > 0.2f ? 1.0f : 0.0f;

  // position-major head outputs: [b][l][64]
  const long rb = (long)b * 16384 + (long)l * 64;
  const float4 r = *(const float4*)(oreg + rb + j * 4);
  const float px = ax + r.x;
  const float py = ay + r.y;
  const float pw = aw + r.z;
  const float ph = ah + r.w;
  const float z  = ocls[rb + j];

  const float lc = fabsf(px - gx) + fabsf(py - gy);
  const float lh = fabsf(pw - gw) + fabsf(ph - gh);
  float piou, pg;
  iou_giou_dev(px - 0.5f * pw, py - 0.5f * ph, px + 0.5f * pw, py + 0.5f * ph,
               gx1, gy1, gx2, gy2, piou, pg);
  const float lg  = 1.0f - pg;
  const float bce = fmaxf(z, 0.f) - z * pos + log1pf(expf(-fabsf(z)));

  float vals[5] = {pos, pos * lc, pos * lh, pos * lg, bce};
#pragma unroll
  for (int off = 16; off > 0; off >>= 1)
#pragma unroll
    for (int k = 0; k < 5; ++k)
      vals[k] += __shfl_down(vals[k], off, 32);
  if ((threadIdx.x & 31) == 0)
#pragma unroll
    for (int k = 0; k < 5; ++k)
      atomicAdd(&sums[k], vals[k]);
}

__global__ void zero_sums(float* s) { if (threadIdx.x < 8) s[threadIdx.x] = 0.0f; }

__global__ void finalize_loss(const float* __restrict__ s, float* __restrict__ out)
{
  float npos = fmaxf(s[0], 1.0f);
  out[0] = (s[1] + s[2] + 0.3f * s[3]) / npos + 100.0f * (s[4] / 737280.0f);
}

// ---------------------------------------------------------------------------
// Host
// ---------------------------------------------------------------------------
extern "C" void kernel_launch(void* const* d_in, const int* in_sizes, int n_in,
                              void* d_out, int out_size, void* d_ws, size_t ws_size,
                              hipStream_t stream)
{
  const float* feat = (const float*)d_in[0];
  const float* gt   = (const float*)d_in[1];

  // Layer order: 0 in_conv, 1-3 reg_convs, 4-6 cls_convs, 7 reg_head, 8 cls_head
  int wi[9], bi[9], gi[9], bti[9], rmi[9], rvi[9];
  const bool insertion = (n_in > 2 && in_sizes[2] == 131072);
  if (insertion) {
    // dict insertion order: each bn block is (w,b,g,bt,rm,rv)
    const int base[7] = {2, 8, 14, 20, 26, 32, 38};
    for (int L = 0; L < 7; ++L) {
      wi[L] = base[L]; bi[L] = base[L] + 1; gi[L] = base[L] + 2;
      bti[L] = base[L] + 3; rmi[L] = base[L] + 4; rvi[L] = base[L] + 5;
    }
    wi[7] = 44; bi[7] = 45; wi[8] = 46; bi[8] = 47;
  } else {
    // jax tree_flatten sorted-key order: bn blocks are (b,bt,g,rm,rv,w);
    // params keys sorted: cls_convs(2..19), cls_head(20,21), in_conv(22..27),
    // reg_convs(28..45), reg_head(46,47)
    const int base[7] = {22, 28, 34, 40, 2, 8, 14};
    for (int L = 0; L < 7; ++L) {
      bi[L] = base[L]; bti[L] = base[L] + 1; gi[L] = base[L] + 2;
      rmi[L] = base[L] + 3; rvi[L] = base[L] + 4; wi[L] = base[L] + 5;
    }
    bi[7] = 46; wi[7] = 47; bi[8] = 20; wi[8] = 21;
  }
  for (int L = 7; L < 9; ++L) { gi[L] = bti[L] = rmi[L] = rvi[L] = -1; }

  const int Nact[9]   = {512, 256, 256, 256, 256, 256, 256, 48, 12};
  const int Npad[9]   = {512, 256, 256, 256, 256, 256, 256, 64, 64};
  const int rowOff[9] = {0, 512, 768, 1024, 1280, 1536, 1792, 2048, 2112};
  const int totRows   = 2176;

  // workspace layout
  char* ws = (char*)d_ws;
  size_t off = 0;
  float* sums = (float*)(ws + off); off = 256;
  unsigned* W2 = (unsigned*)(ws + off); off += (size_t)totRows * KDW * 4;
  off = (off + 255) & ~(size_t)255;
  float* biasw = (float*)(ws + off); off += (size_t)totRows * 4;
  off = (off + 255) & ~(size_t)255;
  unsigned* Y    = (unsigned*)(ws + off); off += (size_t)240 * 256 * 256 * 4; // [b][256l][256c2]
  unsigned* bufA = (unsigned*)(ws + off); off += (size_t)240 * 256 * 128 * 4; // [b][256l][128c2]
  unsigned* bufB = (unsigned*)(ws + off); off += (size_t)240 * 256 * 128 * 4;
  float* oreg = (float*)(ws + off); off += (size_t)240 * 256 * 64 * 4;        // [b][256l][64o]
  float* ocls = (float*)(ws + off); off += (size_t)240 * 256 * 64 * 4;
  (void)ws_size; (void)out_size;

  zero_sums<<<1, 32, 0, stream>>>(sums);

  for (int L = 0; L < 9; ++L) {
    const int total = Npad[L] * KDW;
    prep_weights<<<(total + 255) / 256, 256, 0, stream>>>(
        (const float*)d_in[wi[L]], (const float*)d_in[bi[L]],
        gi[L]  >= 0 ? (const float*)d_in[gi[L]]  : nullptr,
        bti[L] >= 0 ? (const float*)d_in[bti[L]] : nullptr,
        rmi[L] >= 0 ? (const float*)d_in[rmi[L]] : nullptr,
        rvi[L] >= 0 ? (const float*)d_in[rvi[L]] : nullptr,
        Nact[L], Npad[L], W2 + (long)rowOff[L] * KDW, biasw + rowOff[L]);
  }

  // feat f32 planar -> packed position-major (staged in bufB, dead until L2)
  convert_feat<<<dim3(4, 4, 240), dim3(256), 0, stream>>>(feat, bufB);

  const dim3 blk(128);
  // L0: bufB(packed feat) -> Y packed [b][l][256 c2]
  gemm_bn_lrelu<false><<<dim3(480, 8), blk, 0, stream>>>(
      bufB, 32768L, 128, W2 + (long)rowOff[0] * KDW, biasw + rowOff[0],
      (void*)Y, 65536L, 256);
  // reg chain: Y cols [0,128) -> bufA -> bufB -> bufA
  gemm_bn_lrelu<false><<<dim3(480, 4), blk, 0, stream>>>(
      Y, 65536L, 256, W2 + (long)rowOff[1] * KDW, biasw + rowOff[1],
      (void*)bufA, 32768L, 128);
  gemm_bn_lrelu<false><<<dim3(480, 4), blk, 0, stream>>>(
      bufA, 32768L, 128, W2 + (long)rowOff[2] * KDW, biasw + rowOff[2],
      (void*)bufB, 32768L, 128);
  gemm_bn_lrelu<false><<<dim3(480, 4), blk, 0, stream>>>(
      bufB, 32768L, 128, W2 + (long)rowOff[3] * KDW, biasw + rowOff[3],
      (void*)bufA, 32768L, 128);
  // reg head: bufA -> oreg ([b][l][64] f32)
  gemm_bn_lrelu<true><<<dim3(480, 1), blk, 0, stream>>>(
      bufA, 32768L, 128, W2 + (long)rowOff[7] * KDW, biasw + rowOff[7],
      (void*)oreg, 16384L, 64);
  // cls chain: Y cols [128,256) -> bufB -> Y cols [0,128) (reused) -> bufB
  gemm_bn_lrelu<false><<<dim3(480, 4), blk, 0, stream>>>(
      Y + 128, 65536L, 256, W2 + (long)rowOff[4] * KDW, biasw + rowOff[4],
      (void*)bufB, 32768L, 128);
  gemm_bn_lrelu<false><<<dim3(480, 4), blk, 0, stream>>>(
      bufB, 32768L, 128, W2 + (long)rowOff[5] * KDW, biasw + rowOff[5],
      (void*)Y, 65536L, 256);
  gemm_bn_lrelu<false><<<dim3(480, 4), blk, 0, stream>>>(
      Y, 65536L, 256, W2 + (long)rowOff[6] * KDW, biasw + rowOff[6],
      (void*)bufB, 32768L, 128);
  // cls head: bufB -> ocls ([b][l][64] f32)
  gemm_bn_lrelu<true><<<dim3(480, 1), blk, 0, stream>>>(
      bufB, 32768L, 128, W2 + (long)rowOff[8] * KDW, biasw + rowOff[8],
      (void*)ocls, 16384L, 64);

  loss_kernel<<<dim3(2880), dim3(256), 0, stream>>>(oreg, ocls, gt, sums);
  finalize_loss<<<1, 1, 0, stream>>>(sums, (float*)d_out);
}
